// GraphPool_60129542144694
// MI455X (gfx1250) — matrix operand order repaired and checked by reading, not compiled, not run on hardware
//
#include <hip/hip_runtime.h>
#include <stdint.h>
#include <stddef.h>

// GraphPool for MI455X (gfx1250, wave32):
//  K0 init          : zero d_out / flags / counts / scalars, fill hash with empty keys
//  K1 mark          : presence flags for cluster ids
//  K2/K3 scan       : hierarchical prefix sum -> contiguous relabel ranks, nclus
//  K4 inv           : inv[i] = rank[cluster[i]], counts via u32 atomics
//  K5 wconv         : W fp32 -> f16 staging copy in workspace
//  K6 gemm_scatter  : TDM-load W tile to LDS, async-load inv ids to LDS,
//                     f16 WMMA (f32 acc), scatter-add rows into x_pooled with
//                     global_atomic_add_f32
//  K7 normalize     : divide by counts, add bias
//  K8 edge_insert   : relabel edges, dedupe via u64 hash set (atomicCAS)
//  K9 emit          : write pooled edge list + inv (as float) after x_pooled

#define DIM 128
#define SCAN_B 512

typedef __attribute__((ext_vector_type(16))) _Float16 v16h;
typedef __attribute__((ext_vector_type(8)))  float    v8f;

// ---------------------------------------------------------------- K0: init
__global__ __launch_bounds__(256) void k_init(float* __restrict__ out, int out_size,
    unsigned* __restrict__ rank, unsigned* __restrict__ counts,
    unsigned* __restrict__ scal, unsigned* __restrict__ bsums,
    unsigned long long* __restrict__ hash, unsigned hcap, int Nn)
{
  size_t i = (size_t)blockIdx.x * 256 + threadIdx.x;
  if (i < (size_t)out_size) out[i] = 0.0f;
  if (i < (size_t)Nn) { rank[i] = 0u; counts[i] = 0u; }
  if (i < 16)   scal[i] = 0u;
  if (i < 1024) bsums[i] = 0u;
  if (i < (size_t)hcap) hash[i] = ~0ull;
}

// ---------------------------------------------------------------- K1: mark present cluster ids
__global__ __launch_bounds__(256) void k_mark(const int* __restrict__ cluster,
                                              unsigned* __restrict__ rank, int Nn)
{
  int i = blockIdx.x * 256 + threadIdx.x;
  if (i < Nn) rank[cluster[i]] = 1u;   // benign races
}

// ---------------------------------------------------------------- K2: per-block inclusive scan
__global__ __launch_bounds__(SCAN_B) void k_scan1(unsigned* __restrict__ rank,
                                                  unsigned* __restrict__ bsums, int Nn)
{
  __shared__ unsigned s[SCAN_B];
  const int t = threadIdx.x;
  const int i = blockIdx.x * SCAN_B + t;
  unsigned v = (i < Nn) ? rank[i] : 0u;
  s[t] = v;
  __syncthreads();
#pragma unroll
  for (int off = 1; off < SCAN_B; off <<= 1) {
    unsigned u = (t >= off) ? s[t - off] : 0u;
    __syncthreads();
    s[t] += u;
    __syncthreads();
  }
  if (i < Nn) rank[i] = s[t];
  if (t == SCAN_B - 1) bsums[blockIdx.x] = s[t];
}

// ---------------------------------------------------------------- K3: scan of block sums -> exclusive offsets + nclus
__global__ __launch_bounds__(SCAN_B) void k_scan2(unsigned* __restrict__ bsums,
                                                  unsigned* __restrict__ scal, int nb)
{
  __shared__ unsigned s[SCAN_B];
  const int t = threadIdx.x;
  unsigned v = (t < nb) ? bsums[t] : 0u;
  s[t] = v;
  __syncthreads();
#pragma unroll
  for (int off = 1; off < SCAN_B; off <<= 1) {
    unsigned u = (t >= off) ? s[t - off] : 0u;
    __syncthreads();
    s[t] += u;
    __syncthreads();
  }
  if (t < nb) bsums[t] = s[t] - v;          // exclusive
  if (t == SCAN_B - 1) scal[0] = s[t];      // nclus
}

// ---------------------------------------------------------------- K4: inv + counts
__global__ __launch_bounds__(256) void k_inv(const int* __restrict__ cluster,
    const unsigned* __restrict__ rank, const unsigned* __restrict__ bsums,
    unsigned* __restrict__ invp, unsigned* __restrict__ counts, int Nn)
{
  int i = blockIdx.x * 256 + threadIdx.x;
  if (i >= Nn) return;
  int c = cluster[i];
  unsigned iv = rank[c] + bsums[c / SCAN_B] - 1u;
  invp[i] = iv;
  atomicAdd(&counts[iv], 1u);
}

// ---------------------------------------------------------------- K5: W fp32 -> f16
__global__ __launch_bounds__(256) void k_wconv(const float* __restrict__ W,
                                               _Float16* __restrict__ Wh, int n)
{
  int i = blockIdx.x * 256 + threadIdx.x;
  if (i < n) Wh[i] = (_Float16)W[i];
}

// ---------------------------------------------------------------- K6: WMMA GEMM + atomic scatter
__global__ __launch_bounds__(256) void k_gemm_scatter(const float* __restrict__ x,
    const _Float16* __restrict__ Wh, const unsigned* __restrict__ invp,
    const unsigned* __restrict__ scal, float* __restrict__ out, int Nn)
{
  __shared__ __align__(128) _Float16 sW[DIM * DIM];   // 32 KB f16 W tile
  __shared__ __align__(64) unsigned  sInv[128];       // this block's cluster ids
  const int tid  = threadIdx.x;
  const int wave = tid >> 5;
  const int lane = tid & 31;
  const int hi   = lane >> 4;    // half-wave selector
  const int r    = lane & 15;
  const int b    = blockIdx.y;
  const unsigned nclus = scal[0];
  const int blockBase = blockIdx.x * 128;

  // -------- stage this block's 128 inv[] ids into LDS with async-to-LDS
#if defined(__AMDGCN__) && __has_builtin(__builtin_amdgcn_global_load_async_to_lds_b32)
  if (tid < 128) {
    int idx = blockBase + tid;
    if (idx >= Nn) idx = Nn - 1;                      // clamp; OOB rows never scattered
    __attribute__((address_space(1))) int* g =
        (__attribute__((address_space(1))) int*)(uintptr_t)(invp + idx);
    __attribute__((address_space(3))) int* l =
        (__attribute__((address_space(3))) int*)(unsigned)(uintptr_t)(&sInv[tid]);
    __builtin_amdgcn_global_load_async_to_lds_b32(g, l, 0, 0);
  }
#else
  if (tid < 128) {
    int idx = blockBase + tid;
    if (idx >= Nn) idx = Nn - 1;
    sInv[tid] = invp[idx];
  }
#endif

  // -------- stage W into LDS: TDM preferred, async-to-LDS fallback, plain fallback
#if defined(__AMDGCN__) && __has_builtin(__builtin_amdgcn_tensor_load_to_lds) && __has_builtin(__builtin_amdgcn_s_wait_tensorcnt)
  if (wave == 0) {
    typedef __attribute__((ext_vector_type(4))) unsigned int u32x4;
    typedef __attribute__((ext_vector_type(8))) int i32x8;
    typedef __attribute__((ext_vector_type(4))) int i32x4;
    unsigned long long ga = (unsigned long long)(uintptr_t)Wh;
    unsigned lds = (unsigned)(uintptr_t)(&sW[0]);
    // D# group0: count=1, lds_addr, global_addr[56:0], type=2
    u32x4 g0;
    g0[0] = 1u;
    g0[1] = lds;
    g0[2] = (unsigned)(ga & 0xFFFFFFFFu);
    g0[3] = (unsigned)((ga >> 32) & 0x1FFFFFFu) | (2u << 30);
    // D# group1: data_size=2B, tensor 128x128, tile 128x128, stride0=128, stride1=16384
    i32x8 g1;
    g1[0] = (int)(1u << 16);            // data_size=1 (2 bytes)
    g1[1] = (int)(128u << 16);          // tensor_dim0 low16 @ [31:16]
    g1[2] = (int)(128u << 16);          // tensor_dim0 hi=0, tensor_dim1 low16 @ [31:16]
    g1[3] = (int)(128u << 16);          // tensor_dim1 hi=0, tile_dim0=128 @ [31:16]
    g1[4] = 128;                        // tile_dim1=128, tile_dim2=0
    g1[5] = 128;                        // tensor_dim0_stride low32
    g1[6] = (int)(16384u << 16);        // stride0 hi=0, tensor_dim1_stride low16
    g1[7] = 0;                          // tensor_dim1_stride hi
    i32x4 gz = {0, 0, 0, 0};
#if defined(__clang_major__) && (__clang_major__ >= 23)
    i32x8 gz8 = {0, 0, 0, 0, 0, 0, 0, 0};
    __builtin_amdgcn_tensor_load_to_lds(g0, g1, gz, gz, gz8, 0);
#else
    __builtin_amdgcn_tensor_load_to_lds(g0, g1, gz, gz, 0);
#endif
    __builtin_amdgcn_s_wait_tensorcnt(0);
  }
#elif defined(__AMDGCN__) && __has_builtin(__builtin_amdgcn_global_load_async_to_lds_b128)
  {
    for (int i = tid; i < (DIM * DIM * 2) / 16; i += 256) {
      __attribute__((address_space(1))) int* g =
          (__attribute__((address_space(1))) int*)((uintptr_t)Wh + (size_t)i * 16);
      __attribute__((address_space(3))) int* l =
          (__attribute__((address_space(3))) int*)((unsigned)(uintptr_t)(&sW[0]) + i * 16);
      __builtin_amdgcn_global_load_async_to_lds_b128(g, l, 0, 0);
    }
  }
#else
  {
    const uint4* src = (const uint4*)(const void*)Wh;
    uint4* dst = (uint4*)(void*)&sW[0];
    for (int i = tid; i < (DIM * DIM * 2) / 16; i += 256) dst[i] = src[i];
  }
#endif

  // drain async LDS traffic before the barrier (ASYNCcnt)
#if defined(__AMDGCN__)
#if __has_builtin(__builtin_amdgcn_s_wait_asynccnt)
  __builtin_amdgcn_s_wait_asynccnt(0);
#else
  asm volatile("s_wait_asynccnt 0" ::: "memory");
#endif
#endif
  __syncthreads();

  // -------- load A fragments: 16 rows x 128 K, f32 -> f16 on the fly
  const int base = blockBase + wave * 16;
  int row = base + r;
  if (row >= Nn) row = Nn - 1;                 // clamp; OOB rows never scattered
  const float* xrow = x + ((size_t)b * (size_t)Nn + (size_t)row) * DIM;
  __builtin_prefetch(xrow, 0, 1);

  v16h A[4];
#pragma unroll
  for (int k = 0; k < 4; ++k) {
    // 16-bit A layout: half-wave hi covers K = {k*32 + hi*8 .. +8, k*32+16+hi*8 .. +8}
    const float* p0 = xrow + k * 32 + hi * 8;
    const float* p1 = xrow + k * 32 + 16 + hi * 8;
    float4 q0 = ((const float4*)p0)[0];
    float4 q1 = ((const float4*)p0)[1];
    float4 q2 = ((const float4*)p1)[0];
    float4 q3 = ((const float4*)p1)[1];
    v16h a;
    a[0]=(_Float16)q0.x; a[1]=(_Float16)q0.y; a[2]=(_Float16)q0.z; a[3]=(_Float16)q0.w;
    a[4]=(_Float16)q1.x; a[5]=(_Float16)q1.y; a[6]=(_Float16)q1.z; a[7]=(_Float16)q1.w;
    a[8]=(_Float16)q2.x; a[9]=(_Float16)q2.y; a[10]=(_Float16)q2.z; a[11]=(_Float16)q2.w;
    a[12]=(_Float16)q3.x; a[13]=(_Float16)q3.y; a[14]=(_Float16)q3.z; a[15]=(_Float16)q3.w;
    A[k] = a;
  }

  // cluster ids for the 8 output rows this lane's accumulator components map to
  unsigned myinv[8];
  int rowok[8];
#pragma unroll
  for (int v2 = 0; v2 < 8; ++v2) {
    int rw = wave * 16 + hi * 8 + v2;          // D layout: M = v + 8*hi
    rowok[v2] = (blockBase + rw < Nn);
    myinv[v2] = sInv[rw];
  }

#pragma unroll
  for (int c = 0; c < 8; ++c) {
    const _Float16* wrow = &sW[(c * 16 + r) * DIM];   // B: lane holds column o=r
    v8f acc = {};
#pragma unroll
    for (int k = 0; k < 4; ++k) {
      const _Float16* wp = wrow + k * 32 + hi * 8;   // B: half-wave hi covers K {hi*8..+8, 16+hi*8..+8}
      v16h bt;
#pragma unroll
      for (int t = 0; t < 16; ++t) bt[t] = wp[t < 8 ? t : t + 8];
      acc = __builtin_amdgcn_wmma_f32_16x16x32_f16(
          false, A[k], false, bt, (short)0, acc, false, false);
    }
    const int col = c * 16 + r;
#pragma unroll
    for (int v2 = 0; v2 < 8; ++v2) {
      if (rowok[v2]) {
        float* dst = out + ((size_t)b * (size_t)nclus + (size_t)myinv[v2]) * DIM + col;
        unsafeAtomicAdd(dst, acc[v2]);          // global_atomic_add_f32
      }
    }
  }
}

// ---------------------------------------------------------------- K7: mean + bias
__global__ __launch_bounds__(256) void k_norm(float* __restrict__ out,
    const unsigned* __restrict__ counts, const float* __restrict__ bias,
    const unsigned* __restrict__ scal, int Bb, int out_size)
{
  unsigned nclus = scal[0];
  size_t total = (size_t)Bb * (size_t)nclus * DIM;
  size_t idx = (size_t)blockIdx.x * 256 + threadIdx.x;
  if (idx >= total || idx >= (size_t)out_size) return;
  unsigned o = (unsigned)(idx % DIM);
  unsigned c = (unsigned)((idx / DIM) % nclus);
  out[idx] = out[idx] / (float)counts[c] + bias[o];
}

// ---------------------------------------------------------------- K8: edge relabel + hash dedupe
__global__ __launch_bounds__(256) void k_edge_insert(const int* __restrict__ edges,
    const unsigned* __restrict__ invp, const unsigned* __restrict__ scal,
    unsigned long long* __restrict__ hash, unsigned long long* __restrict__ elist,
    unsigned* __restrict__ ecount, unsigned hmask, int Ee)
{
  int e = blockIdx.x * 256 + threadIdx.x;
  if (e >= Ee) return;
  unsigned nclus = scal[0];
  unsigned a  = invp[edges[e]];
  unsigned bb = invp[edges[Ee + e]];
  unsigned long long key = (unsigned long long)a * (unsigned long long)nclus
                         + (unsigned long long)bb;
  unsigned long long h = key * 0x9E3779B97F4A7C15ull;
  unsigned slot = (unsigned)(h >> 32) & hmask;
  for (unsigned iter = 0; iter <= hmask; ++iter) {
    unsigned long long prev = atomicCAS(&hash[slot], ~0ull, key);
    if (prev == ~0ull) { unsigned p = atomicAdd(ecount, 1u); elist[p] = key; break; }
    if (prev == key) break;
    slot = (slot + 1u) & hmask;
  }
}

// ---------------------------------------------------------------- K9: emit edges + inv
__global__ __launch_bounds__(256) void k_emit(const unsigned long long* __restrict__ elist,
    const unsigned* __restrict__ scal, const unsigned* __restrict__ invp,
    float* __restrict__ out, int Bb, int Nn, int out_size)
{
  int i = blockIdx.x * 256 + threadIdx.x;
  unsigned nclus = scal[0];
  unsigned Eu    = scal[1];
  size_t ebase = (size_t)Bb * (size_t)nclus * DIM;
  if (i < (int)Eu) {
    unsigned long long key = elist[i];
    size_t i0 = ebase + (size_t)i;
    size_t i1 = ebase + (size_t)Eu + (size_t)i;
    if (i1 < (size_t)out_size) {
      out[i0] = (float)(key / nclus);
      out[i1] = (float)(key % nclus);
    }
  }
  if (i < Nn) {
    size_t iv = ebase + 2 * (size_t)Eu + (size_t)i;
    if (iv < (size_t)out_size) out[iv] = (float)invp[i];
  }
}

// ---------------------------------------------------------------- launcher
extern "C" void kernel_launch(void* const* d_in, const int* in_sizes, int n_in,
                              void* d_out, int out_size, void* d_ws, size_t ws_size,
                              hipStream_t stream)
{
  const float* x       = (const float*)d_in[0];
  const int*   edges   = (const int*)d_in[1];
  const int*   cluster = (const int*)d_in[2];
  const float* W       = (const float*)d_in[3];
  const float* bias    = (const float*)d_in[4];

  const int Nn = in_sizes[2];
  const int Ee = in_sizes[1] / 2;
  const int Bb = in_sizes[0] / (Nn * DIM);

  char* ws = (char*)d_ws;
  size_t off = 0;
  auto carve = [&](size_t bytes) -> char* {
    char* p = ws + off;
    off = (off + bytes + 255) & ~(size_t)255;
    return p;
  };
  unsigned* rank   = (unsigned*)carve((size_t)Nn * 4);
  unsigned* bsums  = (unsigned*)carve(1024 * 4);
  unsigned* scal   = (unsigned*)carve(64);                 // [0]=nclus [1]=ecount
  unsigned* invp   = (unsigned*)carve((size_t)Nn * 4);
  unsigned* counts = (unsigned*)carve((size_t)Nn * 4);
  _Float16* Wh     = (_Float16*)carve((size_t)DIM * DIM * 2);
  unsigned long long* elist = (unsigned long long*)carve((size_t)Ee * 8);

  size_t remain = (ws_size > off) ? (ws_size - off) : 0;
  unsigned hlog = 16;
  while (hlog < 23 && (((size_t)1 << (hlog + 1)) * 8) <= remain) ++hlog;
  unsigned hcap = 1u << hlog;
  unsigned long long* hash = (unsigned long long*)carve((size_t)hcap * 8);

  {
    size_t maxe = (size_t)out_size;
    if ((size_t)hcap > maxe) maxe = hcap;
    if ((size_t)Nn > maxe) maxe = Nn;
    if (maxe < 1024) maxe = 1024;
    int blocks = (int)((maxe + 255) / 256);
    k_init<<<blocks, 256, 0, stream>>>((float*)d_out, out_size, rank, counts,
                                       scal, bsums, hash, hcap, Nn);
  }
  const int nblk = (Nn + 255) / 256;
  k_mark<<<nblk, 256, 0, stream>>>(cluster, rank, Nn);
  const int nb = (Nn + SCAN_B - 1) / SCAN_B;
  k_scan1<<<nb, SCAN_B, 0, stream>>>(rank, bsums, Nn);
  k_scan2<<<1, SCAN_B, 0, stream>>>(bsums, scal, nb);
  k_inv<<<nblk, 256, 0, stream>>>(cluster, rank, bsums, invp, counts, Nn);
  k_wconv<<<(DIM * DIM + 255) / 256, 256, 0, stream>>>(W, Wh, DIM * DIM);

  dim3 ggrid((Nn + 127) / 128, Bb);
  k_gemm_scatter<<<ggrid, 256, 0, stream>>>(x, Wh, invp, scal, (float*)d_out, Nn);

  {
    size_t tot = (size_t)Bb * (size_t)Nn * DIM;   // upper bound for B*nclus*DIM
    int blocks = (int)((tot + 255) / 256);
    k_norm<<<blocks, 256, 0, stream>>>((float*)d_out, counts, bias, scal, Bb, out_size);
  }
  k_edge_insert<<<(Ee + 255) / 256, 256, 0, stream>>>(edges, invp, scal, hash,
                                                      elist, scal + 1, hcap - 1, Ee);
  {
    int mx = (Ee > Nn) ? Ee : Nn;
    k_emit<<<(mx + 255) / 256, 256, 0, stream>>>(elist, scal, invp,
                                                 (float*)d_out, Bb, Nn, out_size);
  }
}
